// GradSplitter_70789650972920
// MI455X (gfx1250) — matrix operand order repaired and checked by reading, not compiled, not run on hardware
//
#include <hip/hip_runtime.h>

// ---------------------------------------------------------------------------
// GradSplitter fused kernel for MI455X (gfx1250, wave32, WMMA).
//
// Folding: masks and the first head layer are linear, so everything after
// GCN1 collapses to
//   h1  = relu(x @ W1 + b1)                         [N,200]
//   hid = relu(h1 @ Abig + c1)                      [N,64]  (8 modules x 8)
//   out[n,m] = sigmoid(sum_h hid[n,m*8+h]*hw2[m,h] + hb2[m])
// with Abig[k, m*8+h] = m1[m,k] * sum_c W2[k,c]*m2[m,c]*hw1[m,c,h].
// ---------------------------------------------------------------------------

typedef __bf16 v16bf __attribute__((ext_vector_type(16)));
typedef __bf16 v2bf  __attribute__((ext_vector_type(2)));
typedef float  v8f   __attribute__((ext_vector_type(8)));

// pointer types for the async global->LDS builtin (from the clang diagnostic:
// param is 'int __vector(4) __device__ *', i.e. v4i in address_space(1))
typedef int v4i __attribute__((vector_size(16)));
typedef __attribute__((address_space(1))) v4i gas_v4i;
typedef __attribute__((address_space(3))) v4i las_v4i;

union Frag { v16bf v; unsigned int u[8]; };

__device__ __forceinline__ unsigned packbf(float a, float b) {
    v2bf p;
    p[0] = (__bf16)a;          // native v_cvt on gfx1250
    p[1] = (__bf16)b;
    return __builtin_bit_cast(unsigned, p);
}

__device__ __forceinline__ v8f zero8() {
    v8f z;
#pragma unroll
    for (int i = 0; i < 8; ++i) z[i] = 0.f;
    return z;
}

#if __has_builtin(__builtin_amdgcn_global_load_async_to_lds_b128) && \
    __has_builtin(__builtin_amdgcn_s_wait_asynccnt)
#define USE_ASYNC_LDS 1
#else
#define USE_ASYNC_LDS 0
#endif

// Workspace layout (bytes)
#define WS_W1FRAG   0          // bf16 [16 kchunks][13 cblk][512]  = 212992 B
#define WS_ABIG     212992     // bf16 [ 7 kchunks][ 4 cblk][512]  =  28672 B
#define WS_B1PAD    241664     // f32  [208]
#define WS_C1       242496     // f32  [64]
#define WS_HW2      242752     // f32  [64]
#define WS_HB2      243008     // f32  [8]
#define WS_NEED     243040

// ---------------------------------------------------------------------------
// Prep: fold masks + head layer 1 into weights, pre-swizzle into the CDNA5
// 16-bit B-matrix fragment layout (lane L holds col N=L%16; element e -> K =
// (e&7) + 8*(L>=16) + 16*(e>=8)), pad K and N dims with zeros.
// ---------------------------------------------------------------------------
__global__ __launch_bounds__(256) void prep_kernel(
    const float* __restrict__ W1, const float* __restrict__ b1,
    const float* __restrict__ W2, const float* __restrict__ b2,
    const float* __restrict__ mask1, const float* __restrict__ mask2,
    const float* __restrict__ hw1, const float* __restrict__ hb1,
    const float* __restrict__ hw2, const float* __restrict__ hb2,
    __bf16* __restrict__ w1frag, __bf16* __restrict__ abigfrag,
    float* __restrict__ b1pad, float* __restrict__ c1t,
    float* __restrict__ hw2t, float* __restrict__ hb2t)
{
    const int tid = blockIdx.x * 256 + threadIdx.x;
    const int stride = gridDim.x * 256;

    // W1 in B-fragment order, padded to K=512, cols=208
    for (int i = tid; i < 16 * 13 * 512; i += stride) {
        int e   = i & 15;
        int ln  = (i >> 4) & 31;
        int rem = i >> 9;
        int cb  = rem % 13;
        int kc  = rem / 13;
        int K   = kc * 32 + ((ln >= 16) ? 8 : 0) + (e & 7) + ((e & 8) ? 16 : 0);
        int col = cb * 16 + (ln & 15);
        float v = (K < 500 && col < 200) ? W1[K * 200 + col] : 0.f;
        w1frag[i] = (__bf16)v;
    }

    // Collapsed stage-2 matrix in B-fragment order, padded to K=224
    for (int i = tid; i < 7 * 4 * 512; i += stride) {
        int e  = i & 15;
        int ln = (i >> 4) & 31;
        int cb = (i >> 9) & 3;
        int kc = i >> 11;
        int K   = kc * 32 + ((ln >= 16) ? 8 : 0) + (e & 7) + ((e & 8) ? 16 : 0);
        int col = cb * 16 + (ln & 15);
        int m = col >> 3, h = col & 7;
        float v = 0.f;
        if (K < 200 && mask1[m * 200 + K] > 0.f) {
            float s = 0.f;
#pragma unroll
            for (int c = 0; c < 8; ++c)
                if (mask2[m * 8 + c] > 0.f)
                    s += W2[K * 8 + c] * hw1[(m * 8 + c) * 8 + h];
            v = s;
        }
        abigfrag[i] = (__bf16)v;
    }

    for (int i = tid; i < 208; i += stride) b1pad[i] = (i < 200) ? b1[i] : 0.f;

    for (int i = tid; i < 64; i += stride) {
        int m = i >> 3, h = i & 7;
        float s = hb1[i];
#pragma unroll
        for (int c = 0; c < 8; ++c)
            if (mask2[m * 8 + c] > 0.f) s += b2[c] * hw1[(m * 8 + c) * 8 + h];
        c1t[i]  = s;
        hw2t[i] = hw2[i];
    }
    for (int i = tid; i < 8; i += stride) hb2t[i] = hb2[i];
}

// ---------------------------------------------------------------------------
// Fused main kernel: 128 rows/block, 8 wave32s, 16 rows/wave.
// Stage 1: each lane loads exactly its A-fragment K values (no lane
//          exchange!), cvt to bf16, v_wmma_f32_16x16x32_bf16 over 13 N-tiles
//          x 16 K-chunks; x loads software-pipelined one chunk ahead.
// Stage 2: relu(h1+b1) scattered into per-wave LDS strip in A-frag order,
//          x Abig (async-staged into LDS at kernel start), head + sigmoid.
// ---------------------------------------------------------------------------
__global__ __launch_bounds__(256) void fused_kernel(
    const float* __restrict__ x,
    const __bf16* __restrict__ w1frag,
    const __bf16* __restrict__ abigfrag,
    const float* __restrict__ b1pad,
    const float* __restrict__ c1t,
    const float* __restrict__ hw2t,
    const float* __restrict__ hb2t,
    float* __restrict__ out, int N)
{
    __shared__ __bf16 s2a[8 * 3584];   // per-wave h1 strips, A-frag order
    __shared__ __bf16 s2b[7 * 4 * 512]; // stage-2 B fragments (28 KB)

    const int lane = threadIdx.x & 31;
    const int wave = threadIdx.x >> 5;
    const int lid  = lane & 15;
    const int hiH  = lane >> 4;                 // lane-half selector
    const int tile0 = blockIdx.x * 128;

    // ---- kick off async staging of the stage-2 B matrix into LDS --------
    {
        const int t = threadIdx.x;
#pragma unroll
        for (int i = 0; i < 7; ++i) {
            int idx = (t + i * 256) << 3;       // 8 bf16 = 16 B per transfer
#if USE_ASYNC_LDS
            __builtin_amdgcn_global_load_async_to_lds_b128(
                (gas_v4i*)(abigfrag + idx),
                (las_v4i*)(s2b + idx), 0, 0);
#else
            uint4 v = *(const uint4*)(abigfrag + idx);
            *(uint4*)(s2b + idx) = v;
#endif
        }
    }

    int row  = tile0 + wave * 16 + lid;
    int rowC = row < N ? row : N - 1;           // clamp (loads only; stores guarded)
    const float* xrow = x + (long)rowC * 500;

    float b1v[13];
#pragma unroll
    for (int cb = 0; cb < 13; ++cb) b1v[cb] = b1pad[cb * 16 + lid];

    v8f acc[13];
#pragma unroll
    for (int cb = 0; cb < 13; ++cb) acc[cb] = zero8();

    // ---------------- Stage 1: h1 = relu(x @ W1 + b1) --------------------
    // Lane L's A-fragment K offsets within a 32-chunk: {0,4,16,20}+8*hiH.
    float4 f[4];
#pragma unroll
    for (int q = 0; q < 4; ++q) {
        int k = hiH * 8 + (q & 1) * 4 + (q >> 1) * 16;
        f[q] = *(const float4*)(xrow + k);      // chunk 0 always in range
    }

    for (int kc = 0; kc < 16; ++kc) {
        // pack current chunk into the A fragment
        Frag a;
#pragma unroll
        for (int q = 0; q < 4; ++q) {
            a.u[2 * q]     = packbf(f[q].x, f[q].y);
            a.u[2 * q + 1] = packbf(f[q].z, f[q].w);
        }
        // issue next chunk's x loads early (hide HBM latency under WMMAs)
        if (kc < 15) {
            const int kb = (kc + 1) * 32 + hiH * 8;
#pragma unroll
            for (int q = 0; q < 4; ++q) {
                int k = kb + (q & 1) * 4 + (q >> 1) * 16;
                if (k < 500) f[q] = *(const float4*)(xrow + k);
                else { f[q].x = f[q].y = f[q].z = f[q].w = 0.f; }
            }
        }

        const __bf16* wp = w1frag + ((kc * 13) << 9) + (lane << 4);
        __builtin_prefetch(wp + (13 << 9), 0, 0);   // next chunk -> global_prefetch_b8
        Frag bcur, bnext;
        bcur.v = *(const v16bf*)(wp);
#pragma unroll
        for (int cb = 0; cb < 13; ++cb) {
            if (cb < 12) bnext.v = *(const v16bf*)(wp + ((cb + 1) << 9));
            acc[cb] = __builtin_amdgcn_wmma_f32_16x16x32_bf16(
                false, a.v, false, bcur.v, (short)0, acc[cb], false, false);
            if (cb < 12) bcur = bnext;
        }
    }

    // ------- relu+bias, scatter h1 strip into LDS in A-fragment order ----
    __bf16* sw = s2a + wave * 3584;
    // zero-fill K = 208..223 (chunk 6, e8..15 for every fragment lane)
    {
        uint4 z; z.x = z.y = z.z = z.w = 0u;
        *(uint4*)(sw + 6 * 512 + (lane << 4) + 8) = z;
    }
#pragma unroll
    for (int cb = 0; cb < 13; ++cb) {
        int col = cb * 16 + lid;
        int kc2 = col >> 5, kk = col & 31;
        int e   = (kk & 7) + ((kk & 16) ? 8 : 0);
        int hi2 = (kk >> 3) & 1;
        int base = kc2 * 512 + ((hi2 * 16 + hiH * 8) << 4) + e;
#pragma unroll
        for (int g = 0; g < 8; ++g) {
            float v = acc[cb][g] + b1v[cb];
            v = v > 0.f ? v : 0.f;
            sw[base + (g << 4)] = (__bf16)v;
        }
    }

    // make the async-staged B matrix visible to all waves
#if USE_ASYNC_LDS
    __builtin_amdgcn_s_wait_asynccnt(0);
#endif
    __syncthreads();

    // ---------------- Stage 2: hid = relu(h1 @ Abig + c1) ----------------
    v8f acc2[4];
#pragma unroll
    for (int cb = 0; cb < 4; ++cb) acc2[cb] = zero8();

#pragma unroll
    for (int kc2 = 0; kc2 < 7; ++kc2) {
        Frag a2;
        a2.v = *(const v16bf*)(sw + kc2 * 512 + (lane << 4));
#pragma unroll
        for (int cb = 0; cb < 4; ++cb) {
            Frag b2;
            b2.v = *(const v16bf*)(s2b + (((kc2 << 2) + cb) << 9) + (lane << 4));
            acc2[cb] = __builtin_amdgcn_wmma_f32_16x16x32_bf16(
                false, a2.v, false, b2.v, (short)0, acc2[cb], false, false);
        }
    }

    // -------- head: relu, dot-8 over lanes, sigmoid, store [N,8] ---------
#pragma unroll
    for (int cb = 0; cb < 4; ++cb) {
        int col = cb * 16 + lid;
        int m = col >> 3;
        float c1 = c1t[col];
        float w2 = hw2t[col];
        float bb = hb2t[m];
        bool lead = (lane & 7) == 0;
#pragma unroll
        for (int g = 0; g < 8; ++g) {
            float v = acc2[cb][g] + c1;
            v = v > 0.f ? v : 0.f;
            v *= w2;
            v += __shfl_xor(v, 1, 32);
            v += __shfl_xor(v, 2, 32);
            v += __shfl_xor(v, 4, 32);
            if (lead) {
                int ro = tile0 + wave * 16 + g + hiH * 8;
                if (ro < N)
                    out[ro * 8 + m] = 1.f / (1.f + __expf(-(v + bb)));
            }
        }
    }
}

// ---------------------------------------------------------------------------
extern "C" void kernel_launch(void* const* d_in, const int* in_sizes, int n_in,
                              void* d_out, int out_size, void* d_ws, size_t ws_size,
                              hipStream_t stream) {
    (void)n_in; (void)out_size;
    const float* x     = (const float*)d_in[0];
    const float* W1    = (const float*)d_in[1];
    const float* b1    = (const float*)d_in[2];
    const float* W2    = (const float*)d_in[3];
    const float* b2    = (const float*)d_in[4];
    const float* mask1 = (const float*)d_in[5];
    const float* mask2 = (const float*)d_in[6];
    const float* hw1   = (const float*)d_in[7];
    const float* hb1   = (const float*)d_in[8];
    const float* hw2   = (const float*)d_in[9];
    const float* hb2   = (const float*)d_in[10];

    if (ws_size < (size_t)WS_NEED) return;
    char* ws = (char*)d_ws;
    __bf16* w1frag   = (__bf16*)(ws + WS_W1FRAG);
    __bf16* abigfrag = (__bf16*)(ws + WS_ABIG);
    float* b1pad = (float*)(ws + WS_B1PAD);
    float* c1t   = (float*)(ws + WS_C1);
    float* hw2t  = (float*)(ws + WS_HW2);
    float* hb2t  = (float*)(ws + WS_HB2);
    float* out   = (float*)d_out;

    const int N = in_sizes[0] / 500;   // 150000

    prep_kernel<<<64, 256, 0, stream>>>(W1, b1, W2, b2, mask1, mask2,
                                        hw1, hb1, hw2, hb2,
                                        w1frag, abigfrag, b1pad, c1t, hw2t, hb2t);

    fused_kernel<<<(N + 127) / 128, 256, 0, stream>>>(
        x, w1frag, abigfrag, b1pad, c1t, hw2t, hb2t, out, N);
}
